// FourierBlock_68444598829394
// MI455X (gfx1250) — matrix operand order repaired and checked by reading, not compiled, not run on hardware
//
#include <hip/hip_runtime.h>

// FNO block for MI455X (gfx1250): truncated-DFT formulation, all three GEMMs
// run on v_wmma_f32_16x16x32_f16 with fp32 accumulation.
//
// Workspace layout (bytes):
//   Ft  @ 0      : f16[128][4096]  forward basis, row j, col l  (j<64: cos, j>=64: +sin -> XI' = -Im)
//   Gc  @ 1MB    : f16[4096][128]  inverse basis, row l, col j  ((2-d0)cos ; 2sin), 1/L applied in epilogue
//   WRt @ 2MB    : f16[64][128][128]  Re W, [m][o][i], i contiguous (B-operand layout)
//   WIt @ 4MB    : f16[64][128][128]  Im W
//   WIn @ 6MB    : f16[64][128][128]  -Im W   (f16 WMMA has no B-negate; NEG[1:0]=0 required)
//   X2  @ 8MB    : f16[128][4096]  stage-1 out, row j, col r=b*128+i
//   Y1  @ 9MB    : f16[4096][128]  stage-2 out, row r=b*128+o, col j (m / 64+m)
// Total 10 MB.

#define LL    4096
#define CCH   128
#define NMODE 64
#define AST   136   // padded LDS row stride (halves): 272B = 68 DWORDs -> 4-bank skew per row

typedef __attribute__((ext_vector_type(16))) _Float16 v16h;
typedef __attribute__((ext_vector_type(8)))  float    v8f;

union H8 { uint4 u; _Float16 h[8]; };
union P4 { uint2 u; _Float16 h[4]; };

__device__ __forceinline__ v8f wmma_f16(v16h a, v16h b, v8f c) {
  return __builtin_amdgcn_wmma_f32_16x16x32_f16(false, a, false, b, (short)0, c, false, false);
}

// A fragment (16x32, f16) from row-major f16 [row][k], leading dim ld (global).
// ISA layout: lanes 0-15 rows M=lane, K runs {k0+0..7, k0+16..23}; lanes 16-31 runs +8.
__device__ __forceinline__ v16h load_a_h(const _Float16* base, int ld, int row0, int k0, int lane) {
  int lr = lane & 15;
  int lk = (lane >> 4) << 3;
  const _Float16* p = base + (size_t)(row0 + lr) * ld + k0 + lk;
  H8 x0, x1;
  x0.u = *(const uint4*)p;
  x1.u = *(const uint4*)(p + 16);
  v16h a;
#pragma unroll
  for (int h = 0; h < 8; ++h) { a[h] = x0.h[h]; a[8 + h] = x1.h[h]; }
  return a;
}

// A fragment from LDS (row stride AST halves), same layout -> ds_read_b128 x2.
__device__ __forceinline__ v16h lds_a_frag(const _Float16* Abuf, int kt, int lane) {
  int lr = lane & 15;
  int lk = (lane >> 4) << 3;
  const _Float16* p = Abuf + lr * AST + kt + lk;
  H8 x0, x1;
  x0.u = *(const uint4*)p;
  x1.u = *(const uint4*)(p + 16);
  v16h a;
#pragma unroll
  for (int h = 0; h < 8; ++h) { a[h] = x0.h[h]; a[8 + h] = x1.h[h]; }
  return a;
}

// B fragment (32x16, f16). Lane holds B-column n = lane&15, K runs {k0+(lane>>4)*8+0..7, +16..23}.
// Source stored transposed: base[n][k] row-major, so each 8-half K run is contiguous.
__device__ __forceinline__ v16h load_b_h(const _Float16* base, int ld, int n0, int k0, int lane) {
  int n  = lane & 15;
  int kb = (lane >> 4) << 3;
  const _Float16* p = base + (size_t)(n0 + n) * ld + k0 + kb;
  H8 x0, x1;
  x0.u = *(const uint4*)p;
  x1.u = *(const uint4*)(p + 16);
  v16h b;
#pragma unroll
  for (int h = 0; h < 8; ++h) { b[h] = x0.h[h]; b[8 + h] = x1.h[h]; }
  return b;
}

// ---------------- Stage 0: tables -------------------------------------------
__global__ void fno_init(const float* __restrict__ wr, const float* __restrict__ wi,
                         _Float16* __restrict__ Ft, _Float16* __restrict__ Gc,
                         _Float16* __restrict__ WRt, _Float16* __restrict__ WIt,
                         _Float16* __restrict__ WIn) {
  int idx = blockIdx.x * blockDim.x + threadIdx.x;   // 0 .. 64*128*128-1
  // Weight repack: [i][o][m] fp32 -> [m][o][i] f16 (i contiguous = B-operand K runs)
  {
    int i = idx & 127;
    int o = (idx >> 7) & 127;
    int m = idx >> 14;
    size_t s = (size_t)(i * CCH + o) * NMODE + m;
    float r  = wr[s];
    float im = wi[s];
    WRt[idx] = (_Float16)r;
    WIt[idx] = (_Float16)im;
    WIn[idx] = (_Float16)(-im);
  }
  if (idx < CCH * LL) {
    const float twopi_over_L = 6.2831853071795864769f / (float)LL;
    // Ft[j][l]: j<64 -> cos(2*pi*j*l/L); j>=64 -> +sin (XI' = -Im convention)
    {
      int l = idx & (LL - 1);
      int j = idx >> 12;
      int m = j & (NMODE - 1);
      float th = twopi_over_L * (float)((m * l) & (LL - 1));   // exact period reduction
      float s, c;
      __sincosf(th, &s, &c);
      Ft[idx] = (_Float16)((j < NMODE) ? c : s);
    }
    // Gc[l][j]: j<64 -> (m==0?1:2)*cos; j>=64 -> (m==0?0:2)*sin. 1/L applied in epilogue.
    {
      int j = idx & 127;
      int l = idx >> 7;
      int m = j & (NMODE - 1);
      float th = twopi_over_L * (float)((m * l) & (LL - 1));
      float s, c;
      __sincosf(th, &s, &c);
      float coef = (m == 0) ? ((j < NMODE) ? 1.0f : 0.0f) : 2.0f;
      Gc[idx] = (_Float16)(coef * ((j < NMODE) ? c : s));
    }
  }
}

// ---------------- Stage 1: forward truncated DFT ----------------------------
// X2(128 x 4096, transposed) = [x(4096 x 4096) @ Ft^T]^T.
// One block per 16-row M-strip; 8 waves = 8 N-tiles share the A strip via LDS
// (f32 feature converted to f16 once, double-buffered 16x128 chunks).
__global__ void __launch_bounds__(256) fno_fwd(const float* __restrict__ x,
                                               const _Float16* __restrict__ Ft,
                                               _Float16* __restrict__ X2) {
  __shared__ _Float16 As[2][16 * AST];
  const int t    = threadIdx.x;
  const int lane = t & 31;
  const int n0   = (t >> 5) * 16;          // wave id = N tile
  const int row0 = blockIdx.x * 16;        // M strip

  // Cooperative chunk loader: 16 rows x 128 cols f32 -> f16 LDS. 256 thr * 8 floats.
  const int lr0 = t >> 5;                  // rows lr0 and lr0+8
  const int c4  = (t & 31) << 2;           // float4 column
  const float* srow0 = x + (size_t)(row0 + lr0) * LL + c4;
  const float* srow1 = srow0 + (size_t)8 * LL;

#define LOAD_CHUNK(buf, kb)                                                    \
  {                                                                            \
    float4 f0 = *(const float4*)(srow0 + (kb));                                \
    float4 f1 = *(const float4*)(srow1 + (kb));                                \
    P4 p0, p1;                                                                 \
    p0.h[0] = (_Float16)f0.x; p0.h[1] = (_Float16)f0.y;                        \
    p0.h[2] = (_Float16)f0.z; p0.h[3] = (_Float16)f0.w;                        \
    p1.h[0] = (_Float16)f1.x; p1.h[1] = (_Float16)f1.y;                        \
    p1.h[2] = (_Float16)f1.z; p1.h[3] = (_Float16)f1.w;                        \
    *(uint2*)&As[buf][lr0 * AST + c4]       = p0.u;                            \
    *(uint2*)&As[buf][(lr0 + 8) * AST + c4] = p1.u;                            \
  }

  LOAD_CHUNK(0, 0)

  v8f acc0 = {}, acc1 = {};
  const int NCH = LL / 128;                // 32 chunks
  for (int c = 0; c < NCH; ++c) {
    __syncthreads();                       // As[c&1] ready; prev readers of As[(c+1)&1] done
    if (c + 1 < NCH) LOAD_CHUNK((c + 1) & 1, (c + 1) * 128)
    const _Float16* Ab = As[c & 1];
    const int kbase = c * 128;
#pragma unroll
    for (int ks = 0; ks < 4; ++ks) {
      v16h a = lds_a_frag(Ab, ks * 32, lane);
      v16h b = load_b_h(Ft, LL, n0, kbase + ks * 32, lane);
      if (ks & 1) acc1 = wmma_f16(a, b, acc1);
      else        acc0 = wmma_f16(a, b, acc0);
    }
  }
#undef LOAD_CHUNK

  // store transposed: X2[j][r] (contiguous over v -> b128 store)
  int ncol = lane & 15;
  int rb   = row0 + ((lane >> 4) << 3);    // C/D layout: lanes16-31 hold M=8..15
#pragma unroll
  for (int v = 0; v < 8; ++v)
    X2[(size_t)(n0 + ncol) * LL + rb + v] = (_Float16)(acc0[v] + acc1[v]);
}

// ---------------- Stage 2: per-mode complex channel mix ---------------------
// For mode m: OutR = XR*WR + XI'*WI ; OutI' = XI'*WR + XR*(-WI).
__global__ void fno_mix(const _Float16* __restrict__ X2, const _Float16* __restrict__ WRt,
                        const _Float16* __restrict__ WIt, const _Float16* __restrict__ WIn,
                        _Float16* __restrict__ Y1) {
  int wave = (blockIdx.x * blockDim.x + threadIdx.x) >> 5;  // 0..511
  int lane = threadIdx.x & 31;
  int m  = wave >> 3;
  int n0 = (wave & 7) * 16;
  const _Float16* XR = X2 + (size_t)m * LL;            // 32x128, ld=128 (r = b*128+i)
  const _Float16* XI = X2 + (size_t)(NMODE + m) * LL;
  const _Float16* WR = WRt + (size_t)m * CCH * CCH;    // [o][i]
  const _Float16* WI = WIt + (size_t)m * CCH * CCH;
  const _Float16* WN = WIn + (size_t)m * CCH * CCH;
  v8f aR0 = {}, aR1 = {}, aI0 = {}, aI1 = {};
  for (int kt = 0; kt < CCH; kt += 32) {
    v16h ar0 = load_a_h(XR, CCH, 0,  kt, lane);
    v16h ar1 = load_a_h(XR, CCH, 16, kt, lane);
    v16h ai0 = load_a_h(XI, CCH, 0,  kt, lane);
    v16h ai1 = load_a_h(XI, CCH, 16, kt, lane);
    v16h br  = load_b_h(WR, CCH, n0, kt, lane);
    v16h bi  = load_b_h(WI, CCH, n0, kt, lane);
    v16h bn  = load_b_h(WN, CCH, n0, kt, lane);
    aR0 = wmma_f16(ar0, br, aR0);  aR0 = wmma_f16(ai0, bi, aR0);
    aR1 = wmma_f16(ar1, br, aR1);  aR1 = wmma_f16(ai1, bi, aR1);
    aI0 = wmma_f16(ai0, br, aI0);  aI0 = wmma_f16(ar0, bn, aI0);
    aI1 = wmma_f16(ai1, br, aI1);  aI1 = wmma_f16(ar1, bn, aI1);
  }
  // scatter into Y1[r=b*128+o][j], j = m (Re) / 64+m (-Im)
  int ncol = lane & 15;
  int rb   = (lane >> 4) << 3;
#pragma unroll
  for (int v = 0; v < 8; ++v) {
    int b0 = rb + v, b1 = 16 + rb + v;
    int o  = n0 + ncol;
    Y1[(size_t)(b0 * CCH + o) * CCH + m]          = (_Float16)aR0[v];
    Y1[(size_t)(b1 * CCH + o) * CCH + m]          = (_Float16)aR1[v];
    Y1[(size_t)(b0 * CCH + o) * CCH + NMODE + m]  = (_Float16)aI0[v];
    Y1[(size_t)(b1 * CCH + o) * CCH + NMODE + m]  = (_Float16)aI1[v];
  }
}

// ---------------- Stage 3: inverse truncated DFT + residual -----------------
// out(4096 x 4096) = x + (1/L) * Y1(4096 x 128) @ Gc^T.  1 wave = 16 rows x 128 cols.
__global__ void fno_inv(const _Float16* __restrict__ Y1, const _Float16* __restrict__ Gc,
                        const float* __restrict__ x, float* __restrict__ out) {
  int wave = (blockIdx.x * blockDim.x + threadIdx.x) >> 5;  // 0..8191
  int lane = threadIdx.x & 31;
  int row0 = (wave >> 5) * 16;
  int lgrp = wave & 31;                 // 8 N-tiles per wave
  v8f acc[8];
#pragma unroll
  for (int t = 0; t < 8; ++t) acc[t] = (v8f){};
  for (int kt = 0; kt < CCH; kt += 32) {
    v16h a = load_a_h(Y1, CCH, row0, kt, lane);
#pragma unroll
    for (int t = 0; t < 8; ++t) {
      int n0 = (lgrp * 8 + t) * 16;
      v16h b = load_b_h(Gc, CCH, n0, kt, lane);
      acc[t] = wmma_f16(a, b, acc[t]);
    }
  }
  int ncol = lane & 15;
  int rb   = row0 + ((lane >> 4) << 3);
  const float invL = 1.0f / (float)LL;
#pragma unroll
  for (int t = 0; t < 8; ++t) {
    int col = (lgrp * 8 + t) * 16 + ncol;
#pragma unroll
    for (int v = 0; v < 8; ++v) {
      size_t idx = (size_t)(rb + v) * LL + col;
      out[idx] = x[idx] + acc[t][v] * invL;
    }
  }
}

// ---------------- Host launcher ---------------------------------------------
extern "C" void kernel_launch(void* const* d_in, const int* in_sizes, int n_in,
                              void* d_out, int out_size, void* d_ws, size_t ws_size,
                              hipStream_t stream) {
  (void)in_sizes; (void)n_in; (void)out_size; (void)ws_size;
  const float* feature = (const float*)d_in[0];
  const float* wr      = (const float*)d_in[1];
  const float* wi      = (const float*)d_in[2];
  float*       out     = (float*)d_out;
  char*        ws      = (char*)d_ws;

  const size_t MB = 1u << 20;
  _Float16* Ft  = (_Float16*)(ws + 0 * MB);
  _Float16* Gc  = (_Float16*)(ws + 1 * MB);
  _Float16* WRt = (_Float16*)(ws + 2 * MB);
  _Float16* WIt = (_Float16*)(ws + 4 * MB);
  _Float16* WIn = (_Float16*)(ws + 6 * MB);
  _Float16* X2  = (_Float16*)(ws + 8 * MB);
  _Float16* Y1  = (_Float16*)(ws + 9 * MB);

  fno_init<<<4096, 256, 0, stream>>>(wr, wi, Ft, Gc, WRt, WIt, WIn);        // 1M threads
  fno_fwd <<<256,  256, 0, stream>>>(feature, Ft, X2);                      // 2048 waves, LDS-shared A
  fno_mix <<<64,   256, 0, stream>>>(X2, WRt, WIt, WIn, Y1);                // 512 waves
  fno_inv <<<1024, 256, 0, stream>>>(Y1, Gc, feature, out);                 // 8192 waves
}
